// PGLU_13340168421983
// MI455X (gfx1250) — matrix operand order repaired and checked by reading, not compiled, not run on hardware
//
#include <hip/hip_runtime.h>
#include <hip/hip_bf16.h>

// ---------------------------------------------------------------------------
// PGLU (potential-gated GRU) on MI455X/gfx1250: 4 bf16 WMMA GEMMs with fused
// epilogues. D = A x B + C via V_WMMA_F32_16X16X32_BF16 (wave32).
// Tile staging uses gfx1250 GLOBAL_LOAD_ASYNC_TO_LDS_B128 (ASYNCcnt) with
// double-buffered LDS: async fetch of tile k+1 overlaps WMMA on tile k.
// ---------------------------------------------------------------------------

typedef __bf16 bf16_t;
typedef __attribute__((ext_vector_type(4)))  bf16_t v4bf;
typedef __attribute__((ext_vector_type(8)))  bf16_t v8bf;
typedef __attribute__((ext_vector_type(16))) bf16_t v16bf;
typedef __attribute__((ext_vector_type(8)))  float  v8f;

#define BB  4096   // batch
#define DIN 2048   // input dim
#define HH  2048   // hidden dim

static __device__ __forceinline__ bf16_t f2bf(float f) {
  // round-to-nearest-even f32 -> bf16
  unsigned u = __builtin_bit_cast(unsigned, f);
  unsigned r = u + 0x7FFFu + ((u >> 16) & 1u);
  unsigned short h = (unsigned short)(r >> 16);
  return __builtin_bit_cast(bf16_t, h);
}

// gfx1250 async global->LDS copy, 16 bytes per lane. VDST = LDS byte offset,
// VADDR = 64-bit global address. Tracked by ASYNCcnt.
static __device__ __forceinline__ void async_copy_b128(unsigned lds_off,
                                                       unsigned long long gaddr) {
  asm volatile("global_load_async_to_lds_b128 %0, %1, off"
               :: "v"(lds_off), "v"(gaddr)
               : "memory");
}
static __device__ __forceinline__ void wait_async0() {
  asm volatile("s_wait_asynccnt 0x0" ::: "memory");
}

// ------------------------- f32 -> bf16 conversion --------------------------
__global__ __launch_bounds__(256) void k_f32_to_bf16(
    const float* __restrict__ in, bf16_t* __restrict__ out, int n) {
  int i = (blockIdx.x * 256 + threadIdx.x) * 4;
  if (i < n) {
    float4 v = *(const float4*)(in + i);
    v4bf o = { f2bf(v.x), f2bf(v.y), f2bf(v.z), f2bf(v.w) };
    *(v4bf*)(out + i) = o;
  }
}

// ------------------------------ WMMA GEMM ----------------------------------
// out[m,n] = epilogue( sum_k A[m,k] * Bw[n,k] + bias[n] )
// A split into two row-major halves A0 (width K0) and A1 (width Ktot-K0) so
// the reference's concat() is implicit. Bw is [N, Ktot] row-major (B-fragment
// friendly: a column of B is contiguous in k).
// Block tile 128x128, 256 threads = 8 waves (2M x 4N), wave tile 64x32,
// K-step 32. LDS rows padded to 40 elems (80 B, multiple of 16 B).
template <int EPI>
__global__ __launch_bounds__(256) void k_gemm_bf16(
    const bf16_t* __restrict__ A0, const bf16_t* __restrict__ A1,
    int K0, int Ktot,
    const bf16_t* __restrict__ Bw,
    const float* __restrict__ bias,
    int M, int N,
    const float* __restrict__ pot_prev,  // EPI 0
    const float* __restrict__ tresh,     // EPI 0
    const float* __restrict__ decay,     // EPI 0
    const float* __restrict__ h_prev,    // EPI 1, 3
    const float* __restrict__ zbuf,      // EPI 3
    bf16_t* __restrict__ out_bf16,       // EPI 0 (activated), EPI 1 (r*h)
    float* __restrict__ out_f32,         // EPI 2 (z), EPI 3 (h_new)
    float* __restrict__ out2)            // EPI 0 (potential_non_gated)
{
  constexpr int BLK = 128, KT = 32, LSTR = 40;
  constexpr int TILE_ELEMS = BLK * LSTR;                 // 5120
  constexpr unsigned TILE_BYTES = TILE_ELEMS * 2;        // 10240
  __shared__ bf16_t lds[4 * TILE_ELEMS];                 // 2 bufs x (A,B)

  const int tid  = threadIdx.x;
  const int lane = tid & 31;
  const int wv   = tid >> 5;
  const int wm   = wv >> 2;       // 0..1  (M direction)
  const int wn   = wv & 3;        // 0..3  (N direction)
  const int lr   = lane & 15;     // row/col inside a 16x16 fragment
  const int lh   = lane >> 4;     // lane half

  const int m0 = blockIdx.y * BLK;
  const int n0 = blockIdx.x * BLK;
  const int K1w = Ktot - K0;

  // staging pattern: thread handles rows rowL and rowL+64, one 16B chunk each
  const int rowL = tid >> 2;              // 0..63
  const int kOff = (tid & 3) * 8;         // 0,8,16,24 (elements)

  // LDS byte offsets (flat-pointer low 32 bits == LDS offset, ISA 10.2)
  const unsigned ldsBase = (unsigned)(uintptr_t)&lds[0];
  const unsigned laOff0  = (unsigned)(rowL * LSTR + kOff) * 2u;
  const unsigned laOff1  = (unsigned)((rowL + 64) * LSTR + kOff) * 2u;

  const bf16_t* gb0 = Bw + (size_t)(n0 + rowL) * Ktot + kOff;
  const bf16_t* gb1 = Bw + (size_t)(n0 + rowL + 64) * Ktot + kOff;

  auto issue_tile = [&](int buf, int kt) {
    const unsigned lb = ldsBase + (unsigned)buf * (2u * TILE_BYTES);
    const bf16_t *ga0, *ga1;
    if (kt < K0) {  // K0 is a multiple of KT, so a tile never straddles
      ga0 = A0 + (size_t)(m0 + rowL) * K0 + kt + kOff;
      ga1 = A0 + (size_t)(m0 + rowL + 64) * K0 + kt + kOff;
    } else {
      const int k1 = kt - K0;
      ga0 = A1 + (size_t)(m0 + rowL) * K1w + k1 + kOff;
      ga1 = A1 + (size_t)(m0 + rowL + 64) * K1w + k1 + kOff;
    }
    async_copy_b128(lb + laOff0, (unsigned long long)(uintptr_t)ga0);
    async_copy_b128(lb + laOff1, (unsigned long long)(uintptr_t)ga1);
    async_copy_b128(lb + TILE_BYTES + laOff0, (unsigned long long)(uintptr_t)(gb0 + kt));
    async_copy_b128(lb + TILE_BYTES + laOff1, (unsigned long long)(uintptr_t)(gb1 + kt));
  };

  v8f acc[4][2] = {};
  const int nk = Ktot / KT;

  issue_tile(0, 0);
  for (int ki = 0; ki < nk; ++ki) {
    const int buf = ki & 1;
    // own async copies landed -> barrier -> everyone's copies landed.
    wait_async0();
    __syncthreads();
    // prefetch next tile into the other buffer while we compute this one.
    // (Reads of that buffer from iter ki-1 drained before this barrier.)
    if (ki + 1 < nk) issue_tile(buf ^ 1, (ki + 1) * KT);

    const bf16_t* sA = lds + buf * (2 * TILE_ELEMS);
    const bf16_t* sB = sA + TILE_ELEMS;

    // A fragments (16x32 bf16): lanes 0-15 K 0-7 & 16-23; lanes 16-31 K 8-15 & 24-31
    v16bf afr[4];
    const int ak0 = lh * 8;
#pragma unroll
    for (int i = 0; i < 4; ++i) {
      const bf16_t* p = &sA[(wm * 64 + i * 16 + lr) * LSTR + ak0];
      v8bf lo = *(const v8bf*)p;
      v8bf hi = *(const v8bf*)(p + 16);
      afr[i] = __builtin_shufflevector(lo, hi, 0,1,2,3,4,5,6,7,8,9,10,11,12,13,14,15);
    }
    // B fragments (32x16 bf16): lane = column n, lanes 0-15 K 0-7 & 16-23; lanes 16-31 K 8-15 & 24-31
    v16bf bfr[2];
    const int bk0 = lh * 8;
#pragma unroll
    for (int j = 0; j < 2; ++j) {
      const bf16_t* p = &sB[(wn * 32 + j * 16 + lr) * LSTR + bk0];
      v8bf lo = *(const v8bf*)p;
      v8bf hi = *(const v8bf*)(p + 16);
      bfr[j] = __builtin_shufflevector(lo, hi, 0,1,2,3,4,5,6,7,8,9,10,11,12,13,14,15);
    }

#pragma unroll
    for (int i = 0; i < 4; ++i)
#pragma unroll
      for (int j = 0; j < 2; ++j)
        acc[i][j] = __builtin_amdgcn_wmma_f32_16x16x32_bf16(
            false, afr[i], false, bfr[j], (short)0, acc[i][j], false, false);
  }

  // ---- fused epilogue; C/D layout: vgpr r -> M = r + 8*lh, lane -> N = lr
#pragma unroll
  for (int i = 0; i < 4; ++i) {
    const int mB = m0 + wm * 64 + i * 16 + lh * 8;
#pragma unroll
    for (int j = 0; j < 2; ++j) {
      const int n = n0 + wn * 32 + j * 16 + lr;
#pragma unroll
      for (int r = 0; r < 8; ++r) {
        const int m = mB + r;
        const size_t idx = (size_t)m * N + n;
        float v = acc[i][j][r];
        if constexpr (EPI == 0) {
          float pot  = v + bias[n] + pot_prev[idx];
          bool fire  = pot > tresh[n];              // relu(pot - tresh) > 0
          out_bf16[idx] = f2bf(fire ? pot : 0.0f);  // activated
          out2[idx]     = fire ? 0.0f : pot * decay[n];  // potential_non_gated
        } else if constexpr (EPI == 1) {
          float rr = 1.0f / (1.0f + __expf(-(v + bias[n])));
          out_bf16[idx] = f2bf(rr * h_prev[idx]);   // r * h_prev
        } else if constexpr (EPI == 2) {
          out_f32[idx] = 1.0f / (1.0f + __expf(-(v + bias[n])));  // z
        } else {
          float nn = tanhf(v + bias[n]);
          float zz = zbuf[idx];
          float hp = h_prev[idx];
          out_f32[idx] = (1.0f - zz) * hp + zz * nn;  // h_new
        }
      }
    }
  }
}

// ------------------------------- launcher ----------------------------------
extern "C" void kernel_launch(void* const* d_in, const int* in_sizes, int n_in,
                              void* d_out, int out_size, void* d_ws, size_t ws_size,
                              hipStream_t stream) {
  const float* x        = (const float*)d_in[0];
  const float* h_prev   = (const float*)d_in[1];
  const float* pot_prev = (const float*)d_in[2];
  const float* tresh    = (const float*)d_in[3];
  const float* decay    = (const float*)d_in[4];
  const float* Wi_w = (const float*)d_in[5];
  const float* Wi_b = (const float*)d_in[6];
  const float* Wr_w = (const float*)d_in[7];
  const float* Wr_b = (const float*)d_in[8];
  const float* Wz_w = (const float*)d_in[9];
  const float* Wz_b = (const float*)d_in[10];
  const float* Wc_w = (const float*)d_in[11];
  const float* Wc_b = (const float*)d_in[12];

  float* h_new = (float*)d_out;                       // output 0
  float* png   = (float*)d_out + (size_t)BB * HH;     // output 1

  char* ws = (char*)d_ws;
  size_t off = 0;
  auto carve = [&](size_t bytes) { char* p = ws + off; off += bytes; return p; };
  bf16_t* x_bf   = (bf16_t*)carve((size_t)BB * DIN * 2);
  bf16_t* h_bf   = (bf16_t*)carve((size_t)BB * HH * 2);
  bf16_t* act_bf = (bf16_t*)carve((size_t)BB * HH * 2);
  bf16_t* rh_bf  = (bf16_t*)carve((size_t)BB * HH * 2);
  float*  z_f    = (float*) carve((size_t)BB * HH * 4);
  bf16_t* wi_bf  = (bf16_t*)carve((size_t)HH * DIN * 2);
  bf16_t* wr_bf  = (bf16_t*)carve((size_t)HH * 2 * HH * 2);
  bf16_t* wz_bf  = (bf16_t*)carve((size_t)HH * 2 * HH * 2);
  bf16_t* wc_bf  = (bf16_t*)carve((size_t)HH * 2 * HH * 2);

  auto conv = [&](const float* in, bf16_t* out, size_t n) {
    k_f32_to_bf16<<<(int)(n / (256 * 4)), 256, 0, stream>>>(in, out, (int)n);
  };
  conv(x,      x_bf,  (size_t)BB * DIN);
  conv(h_prev, h_bf,  (size_t)BB * HH);
  conv(Wi_w,   wi_bf, (size_t)HH * DIN);
  conv(Wr_w,   wr_bf, (size_t)HH * 2 * HH);
  conv(Wz_w,   wz_bf, (size_t)HH * 2 * HH);
  conv(Wc_w,   wc_bf, (size_t)HH * 2 * HH);

  dim3 grid(HH / 128, BB / 128), blk(256);
  // 1) potential/gating GEMM: activated + potential_non_gated
  k_gemm_bf16<0><<<grid, blk, 0, stream>>>(
      x_bf, nullptr, DIN, DIN, wi_bf, Wi_b, BB, HH,
      pot_prev, tresh, decay, nullptr, nullptr, act_bf, nullptr, png);
  // 2) r gate -> r * h_prev (bf16)
  k_gemm_bf16<1><<<grid, blk, 0, stream>>>(
      act_bf, h_bf, HH, 2 * HH, wr_bf, Wr_b, BB, HH,
      nullptr, nullptr, nullptr, h_prev, nullptr, rh_bf, nullptr, nullptr);
  // 3) z gate (f32)
  k_gemm_bf16<2><<<grid, blk, 0, stream>>>(
      act_bf, h_bf, HH, 2 * HH, wz_bf, Wz_b, BB, HH,
      nullptr, nullptr, nullptr, nullptr, nullptr, nullptr, z_f, nullptr);
  // 4) candidate + final blend -> h_new
  k_gemm_bf16<3><<<grid, blk, 0, stream>>>(
      act_bf, rh_bf, HH, 2 * HH, wc_bf, Wc_b, BB, HH,
      nullptr, nullptr, nullptr, h_prev, z_f, nullptr, h_new, nullptr);

  (void)in_sizes; (void)n_in; (void)out_size; (void)ws_size;
}
